// GraphTCNEncoder_22076131902076
// MI455X (gfx1250) — matrix operand order repaired and checked by reading, not compiled, not run on hardware
//
#include <hip/hip_runtime.h>
#include <hip/hip_bf16.h>
#include <math.h>

typedef _Float16 half_t;
typedef __attribute__((ext_vector_type(16))) _Float16 v16h;
typedef __attribute__((ext_vector_type(8)))  _Float16 v8h;
typedef __attribute__((ext_vector_type(8)))  float    v8f;

#define BB 64
#define NN 40
#define TT 256
#define FF 4
#define EMB 16
#define CH 64
#define ROWS (BB*NN)          // 2560
#define ACT_ELEMS ((size_t)ROWS*TT*CH)
#define CNT_BN (2560.0f*256.0f)

__device__ __forceinline__ float lrelu(float a) { return a > 0.f ? a : 0.01f*a; }

// ---------------------------------------------------------------------------
// Stage 1a: per-node embeddings  nodes[h] = x@Wn+bn ; s[h] = nodes.Wa
// ---------------------------------------------------------------------------
__global__ __launch_bounds__(256)
void gat_nodes(const float* __restrict__ x,
               const float* __restrict__ Wn0, const float* __restrict__ bn0, const float* __restrict__ Wa0,
               const float* __restrict__ Wn1, const float* __restrict__ bn1, const float* __restrict__ Wa1,
               float* __restrict__ nodes, float* __restrict__ sBuf)
{
  int idx = blockIdx.x*blockDim.x + threadIdx.x;
  if (idx >= BB*TT*NN) return;
  int n = idx % NN;
  int t = (idx / NN) % TT;
  int b = idx / (NN*TT);
  const float* xp = x + (((size_t)b*NN + n)*TT + t)*FF;
  float x0 = xp[0], x1 = xp[1], x2 = xp[2], x3 = xp[3];
  #pragma unroll
  for (int h = 0; h < 2; ++h) {
    const float* Wn = h ? Wn1 : Wn0;
    const float* bn = h ? bn1 : bn0;
    const float* Wa = h ? Wa1 : Wa0;
    float* np = nodes + ((((size_t)h*BB + b)*TT + t)*NN + n)*EMB;
    float s = 0.f;
    #pragma unroll
    for (int d = 0; d < EMB; ++d) {
      float nd = bn[d] + x0*Wn[d] + x1*Wn[EMB+d] + x2*Wn[2*EMB+d] + x3*Wn[3*EMB+d];
      np[d] = nd;
      s += nd * Wa[d];
    }
    sBuf[(((size_t)h*BB + b)*TT + t)*NN + n] = s;
  }
}

// ---------------------------------------------------------------------------
// Stage 1b: attention softmax + aggregate + skip-proj -> x0 f16 [row][t][c]
// spatial term is identically zero, so attn logit = s[r]+s[c]+(bs.Wa+ba)
// ---------------------------------------------------------------------------
__global__ __launch_bounds__(128)
void gat_attn(const float* __restrict__ x,
              const float* __restrict__ nodes, const float* __restrict__ sBuf,
              const float* __restrict__ Wa0, const float* __restrict__ ba0, const float* __restrict__ bs0,
              const float* __restrict__ Wa1, const float* __restrict__ ba1, const float* __restrict__ bs1,
              const float* __restrict__ Wskip, const float* __restrict__ bskip,
              half_t* __restrict__ X0)
{
  int bt = blockIdx.x;
  int t = bt % TT;
  int b = bt / TT;
  __shared__ float sN[2*NN];
  __shared__ float nd[2*NN*EMB];
  int tid = threadIdx.x;
  for (int i = tid; i < 2*NN*EMB; i += 128) {
    int h = i / (NN*EMB); int r = i % (NN*EMB);
    nd[i] = nodes[(((size_t)h*BB + b)*TT + t)*NN*EMB + r];
  }
  for (int i = tid; i < 2*NN; i += 128) {
    int h = i / NN; int n = i % NN;
    sN[i] = sBuf[(((size_t)h*BB + b)*TT + t)*NN + n];
  }
  __syncthreads();
  if (tid < 2*NN) {
    int h = tid / NN, n = tid % NN;
    const float* Wa = h ? Wa1 : Wa0;
    const float* bs = h ? bs1 : bs0;
    float c0 = (h ? ba1 : ba0)[0];
    #pragma unroll
    for (int d = 0; d < EMB; ++d) c0 += bs[d]*Wa[d];
    const float* sh  = sN + h*NN;
    const float* ndh = nd + h*NN*EMB;
    float sn = sh[n];
    float mx = 0.f;                       // diagonal entry (no edge) is 0
    for (int m = 0; m < NN; ++m)
      if (m != n) mx = fmaxf(mx, lrelu(sn + sh[m] + c0));
    float wdiag = __expf(0.f - mx);
    float denom = wdiag;
    float out[EMB];
    #pragma unroll
    for (int d = 0; d < EMB; ++d) out[d] = wdiag * ndh[n*EMB + d];
    for (int m = 0; m < NN; ++m) {
      if (m == n) continue;
      float w = __expf(lrelu(sn + sh[m] + c0) - mx);
      denom += w;
      #pragma unroll
      for (int d = 0; d < EMB; ++d) out[d] += w * ndh[m*EMB + d];
    }
    const float* xp = x + (((size_t)b*NN + n)*TT + t)*FF;
    float x0 = xp[0], x1 = xp[1], x2 = xp[2], x3 = xp[3];
    float rdenom = 1.f/denom;
    v16h ov;
    #pragma unroll
    for (int d = 0; d < EMB; ++d) {
      int c = h*EMB + d;
      float o = lrelu(out[d]*rdenom);
      float xs = bskip[c] + x0*Wskip[c] + x1*Wskip[32+c] + x2*Wskip[64+c] + x3*Wskip[96+c];
      ov[d] = (half_t)(o + xs);
    }
    *(v16h*)(X0 + ((size_t)(b*NN + n)*TT + t)*CH + h*EMB) = ov;
  }
}

// ---------------------------------------------------------------------------
// Weight repack: f32 [co][ci][k] -> f16 [co][k][ci]  (ci contiguous for WMMA A)
// ---------------------------------------------------------------------------
__global__ __launch_bounds__(256)
void repack_w(const float* __restrict__ W, half_t* __restrict__ Wp,
              int cin, int kt, int n)
{
  int idx = blockIdx.x*256 + threadIdx.x;
  if (idx >= n) return;
  int ci = idx % cin;
  int k  = (idx / cin) % kt;
  int m  = idx / (cin*kt);
  Wp[idx] = (half_t)W[((size_t)m*cin + ci)*kt + k];
}

// ---------------------------------------------------------------------------
// WMMA dilated causal conv, channel-innermost activations:
//   Y[t,co] = bias[co] + sum_k,ci Wp[co,k,ci] * X[t+d(k-K+1), ci]
// One wave per (row, 64-t tile); all NCOT co-tiles fused in-wave so every
// B fragment load feeds NCOT WMMAs. All fragment accesses are b128.
// ---------------------------------------------------------------------------
template<int CIN, int KT, int NCOT, int COS>
__global__ __launch_bounds__(32)
void conv_wmma(const half_t* __restrict__ X, const half_t* __restrict__ Wp,
               const float* __restrict__ bias, half_t* __restrict__ Y,
               int dil)
{
  int wid = blockIdx.x;
  int tT = wid & 3;
  int row = wid >> 2;
  int lane = threadIdx.x;
  int laneHi = lane >> 4;
  int nIdx = lane & 15;
  int t0 = tT * 64;
  const half_t* Xrow = X + (size_t)row * TT * CH;
  v8f acc[NCOT][4] = {};
  #pragma unroll
  for (int kk = 0; kk < CIN; kk += 32) {
    #pragma unroll
    for (int k = 0; k < KT; ++k) {
      // A fragments for all co tiles (lane ci = base..base+7, base+16..+23)
      v16h a[NCOT];
      #pragma unroll
      for (int ct = 0; ct < NCOT; ++ct) {
        int m = ct*16 + nIdx;
        const half_t* wrow = Wp + ((size_t)m*KT + k)*CIN + kk + laneHi*8;
        v8h alo = *(const v8h*)(wrow);
        v8h ahi = *(const v8h*)(wrow + 16);
        a[ct] = __builtin_shufflevector(alo, ahi,
                  0,1,2,3,4,5,6,7,8,9,10,11,12,13,14,15);
      }
      int shift = dil * (k - (KT - 1));          // <= 0, causal left pad
      #pragma unroll
      for (int st = 0; st < 4; ++st) {
        int tsrc = t0 + st*16 + nIdx + shift;
        v16h bfrag = {};
        if (tsrc >= 0) {
          const half_t* xb = Xrow + (size_t)tsrc*CH + kk + laneHi*8;
          v8h blo = *(const v8h*)(xb);
          v8h bhi = *(const v8h*)(xb + 16);
          bfrag = __builtin_shufflevector(blo, bhi,
                    0,1,2,3,4,5,6,7,8,9,10,11,12,13,14,15);
        }
        #pragma unroll
        for (int ct = 0; ct < NCOT; ++ct)
          acc[ct][st] = __builtin_amdgcn_wmma_f32_16x16x32_f16(
              false, a[ct], false, bfrag, (short)0, acc[ct][st], false, false);
      }
    }
  }
  #pragma unroll
  for (int ct = 0; ct < NCOT; ++ct) {
    const float* bp = bias + ct*16 + laneHi*8;  // 8 consecutive out channels
    #pragma unroll
    for (int st = 0; st < 4; ++st) {
      int t = t0 + st*16 + nIdx;
      v8h outv;
      #pragma unroll
      for (int v = 0; v < 8; ++v) outv[v] = (half_t)(acc[ct][st][v] + bp[v]);
      *(v8h*)(Y + ((size_t)row*TT + t)*COS + ct*16 + laneHi*8) = outv;
    }
  }
}

// ---------------------------------------------------------------------------
// Gated activation + BatchNorm statistics. Layout [row][t][c], stride-256
// walk keeps each thread on one fixed channel (256 % 64 == 0).
// ---------------------------------------------------------------------------
#define GATE_ITERS 64
__global__ __launch_bounds__(256)
void gate_sums(const half_t* __restrict__ G, const half_t* __restrict__ S,
               half_t* __restrict__ H, float* __restrict__ sums)
{
  int tid = threadIdx.x;
  size_t base = (size_t)blockIdx.x * (256*GATE_ITERS) + tid;
  float s1 = 0.f, s2 = 0.f;
  for (int i = 0; i < GATE_ITERS; ++i) {
    size_t idx = base + (size_t)i*256;
    float g = (float)G[idx], s = (float)S[idx];
    float h = s / (1.f + __expf(-g));
    H[idx] = (half_t)h;
    s1 += h; s2 += h*h;
  }
  __shared__ float r1[256], r2[256];
  r1[tid] = s1; r2[tid] = s2;
  __syncthreads();
  if (tid < 64) {     // channel of thread tid is tid & 63
    float a1 = r1[tid] + r1[tid+64] + r1[tid+128] + r1[tid+192];
    float a2 = r2[tid] + r2[tid+64] + r2[tid+128] + r2[tid+192];
    atomicAdd(&sums[tid],      a1);
    atomicAdd(&sums[64 + tid], a2);
  }
}

__global__ __launch_bounds__(256)
void bn_apply(const half_t* __restrict__ Hin, const float* __restrict__ gamma,
              const float* __restrict__ beta, const float* __restrict__ sums,
              const half_t* __restrict__ skip, half_t* __restrict__ Out)
{
  size_t idx = (size_t)blockIdx.x*256 + threadIdx.x;
  int c = (int)(idx & 63);
  float m   = sums[c] / CNT_BN;
  float var = sums[64 + c] / CNT_BN - m*m;
  float y = gamma[c] * ((float)Hin[idx] - m) * rsqrtf(var + 1e-5f) + beta[c];
  if (skip) y += (float)skip[idx];
  Out[idx] = (half_t)y;
}

// ---------------------------------------------------------------------------
// Attention-head logits (1 out channel) + per-row softmax-weighted mean
// ---------------------------------------------------------------------------
__global__ __launch_bounds__(256)
void att_logits(const half_t* __restrict__ X, const float* __restrict__ Watt,
                const float* __restrict__ batt, float* __restrict__ L)
{
  int idx = blockIdx.x*256 + threadIdx.x;
  const half_t* xr = X + (size_t)idx*CH;   // idx = row*256 + t, channels contig
  float acc = batt[0];
  #pragma unroll
  for (int c = 0; c < CH; ++c) acc += Watt[c] * (float)xr[c];
  L[idx] = acc;
}

__global__ __launch_bounds__(256)
void row_reduce(const half_t* __restrict__ Pred, const float* __restrict__ L,
                float* __restrict__ pa)
{
  int row = blockIdx.x, tid = threadIdx.x;
  __shared__ float red[256];
  __shared__ float wv[256];
  float l = L[(size_t)row*256 + tid];
  red[tid] = l; __syncthreads();
  for (int off = 128; off > 0; off >>= 1) {
    if (tid < off) red[tid] = fmaxf(red[tid], red[tid+off]);
    __syncthreads();
  }
  float mx = red[0];
  float e = __expf(l - mx);
  __syncthreads();
  red[tid] = e; __syncthreads();
  for (int off = 128; off > 0; off >>= 1) {
    if (tid < off) red[tid] += red[tid+off];
    __syncthreads();
  }
  wv[tid] = e / red[0];
  __syncthreads();
  for (int c = 0; c < 48; ++c) {
    float v = (float)Pred[((size_t)row*256 + tid)*48 + c] * wv[tid];
    __syncthreads();
    red[tid] = v; __syncthreads();
    for (int off = 128; off > 0; off >>= 1) {
      if (tid < off) red[tid] += red[tid+off];
      __syncthreads();
    }
    if (tid == 0) pa[(size_t)row*48 + c] = red[0] * (1.f/256.f);
  }
}

__global__ __launch_bounds__(256)
void head_out(const float* __restrict__ pa,
              const float* __restrict__ Wmu, const float* __restrict__ bmu,
              const float* __restrict__ Wsig, const float* __restrict__ bsig,
              float* __restrict__ out)
{
  int idx = blockIdx.x*256 + threadIdx.x;
  if (idx >= ROWS*32) return;
  int o = idx & 31, r = idx >> 5;
  const float* p = pa + (size_t)r*48;
  float mu = bmu[o], lv = bsig[o];
  #pragma unroll
  for (int c = 0; c < 48; ++c) {
    float pv = p[c];
    mu += pv * Wmu[c*32 + o];
    lv += pv * Wsig[c*32 + o];
  }
  out[idx] = mu;
  out[ROWS*32 + idx] = __expf(0.5f*lv);
}

// ---------------------------------------------------------------------------
extern "C" void kernel_launch(void* const* d_in, const int* in_sizes, int n_in,
                              void* d_out, int out_size, void* d_ws, size_t ws_size,
                              hipStream_t stream) {
  (void)in_sizes; (void)n_in; (void)out_size; (void)ws_size;
  auto P = [&](int i) { return (const float*)d_in[i]; };
  // flatten order: inputs, rel_rec, rel_send, then params tree (sorted keys):
  // 3 Watt, 4 Wmu, 5 Wpred, 6 Wsig, 7 Wskip, 8 batt,
  // 9.. blocks x5 of [Wg1,Wg2,Ws1,Ws2,Wsk,b1,b2,bg1,bg2,bs1,bs2,bsk,g1,g2],
  // 79 bmu, 80 bpred, 81 bsig, 82 bskip, 83.. heads x2 of [Wa,Wn,Ws,ba,bn,bs]
  const float* inputs = P(0);

  // workspace carve-up (activations f16, [row][t][c])
  half_t* bufXA = (half_t*)d_ws;
  half_t* bufXB = bufXA + ACT_ELEMS;
  half_t* bufG  = bufXB + ACT_ELEMS;
  half_t* bufS  = bufG  + ACT_ELEMS;
  half_t* bufH  = bufS  + ACT_ELEMS;
  float*  nodes = (float*)(bufH + ACT_ELEMS);               // 2*B*T*N*16
  float*  sBuf  = nodes + (size_t)2*BB*TT*NN*EMB;           // 2*B*T*N
  float*  attL  = sBuf  + (size_t)2*BB*TT*NN;               // ROWS*T
  float*  pa    = attL  + (size_t)ROWS*TT;                  // ROWS*48
  float*  sums  = pa    + (size_t)ROWS*48;                  // 128 floats
  half_t* wpool = (half_t*)(sums + 128);                    // packed weights

  // --- pack all conv weights to f16 [co][k][ci] -----------------------------
  half_t* wcur = wpool;
  auto pack = [&](const float* W, int co, int cin, int kt) -> half_t* {
    half_t* dst = wcur;
    int n = co*cin*kt;
    wcur += n;
    repack_w<<<(n + 255)/256, 256, 0, stream>>>(W, dst, cin, kt, n);
    return dst;
  };
  half_t *pWg1[5], *pWs1[5], *pWg2[5], *pWs2[5], *pWsk[5], *pPred;
  for (int i = 0; i < 5; ++i) {
    int base = 9 + i*14;
    int cin = (i == 0) ? 32 : 64;
    pWg1[i] = pack(P(base+0), 64, cin, 3);
    pWs1[i] = pack(P(base+2), 64, cin, 3);
    pWg2[i] = pack(P(base+1), 64, 64, 3);
    pWs2[i] = pack(P(base+3), 64, 64, 3);
    pWsk[i] = pack(P(base+4), 64, cin, 1);
  }
  pPred = pack(P(5), 48, 64, 1);

  // --- GAT heads + skip projection -> x0 (f16, 32 valid channels of 64) ----
  gat_nodes<<<(BB*TT*NN + 255)/256, 256, 0, stream>>>(
      inputs, P(84), P(87), P(83), P(90), P(93), P(89), nodes, sBuf);
  gat_attn<<<BB*TT, 128, 0, stream>>>(
      inputs, nodes, sBuf, P(83), P(86), P(88), P(89), P(92), P(94),
      P(7), P(82), bufXA);

  auto conv = [&](const half_t* X, const half_t* Wp, const float* b, half_t* Y,
                  int cin, int kt, int dil) {
    dim3 g(ROWS * 4);
    if      (cin == 32 && kt == 3) conv_wmma<32,3,4,64><<<g,32,0,stream>>>(X,Wp,b,Y,dil);
    else if (cin == 64 && kt == 3) conv_wmma<64,3,4,64><<<g,32,0,stream>>>(X,Wp,b,Y,dil);
    else if (cin == 32 && kt == 1) conv_wmma<32,1,4,64><<<g,32,0,stream>>>(X,Wp,b,Y,dil);
    else                           conv_wmma<64,1,4,64><<<g,32,0,stream>>>(X,Wp,b,Y,dil);
  };

  // --- 5 dilated gated TCN blocks ------------------------------------------
  half_t* Xin = bufXA; half_t* Xout = bufXB;
  for (int i = 0; i < 5; ++i) {
    int d = 1 << i;
    int cin = (i == 0) ? 32 : 64;
    int base = 9 + i*14;
    conv(Xin, pWg1[i], P(base+7), bufG, cin, 3, d);           // Wg1, bg1
    conv(Xin, pWs1[i], P(base+9), bufS, cin, 3, d);           // Ws1, bs1
    hipMemsetAsync(sums, 0, 128*sizeof(float), stream);
    gate_sums<<<(int)(ACT_ELEMS/(256*GATE_ITERS)), 256, 0, stream>>>(
        bufG, bufS, bufH, sums);
    bn_apply<<<ROWS*64, 256, 0, stream>>>(bufH, P(base+12), P(base+5), sums,
                                          (const half_t*)nullptr, bufH);
    conv(bufH, pWg2[i], P(base+8),  bufG, 64, 3, d);          // Wg2, bg2
    conv(bufH, pWs2[i], P(base+10), bufS, 64, 3, d);          // Ws2, bs2
    hipMemsetAsync(sums, 0, 128*sizeof(float), stream);
    gate_sums<<<(int)(ACT_ELEMS/(256*GATE_ITERS)), 256, 0, stream>>>(
        bufG, bufS, bufH, sums);
    conv(Xin, pWsk[i], P(base+11), bufS, cin, 1, 1);          // Wsk, bsk
    bn_apply<<<ROWS*64, 256, 0, stream>>>(bufH, P(base+13), P(base+6), sums,
                                          bufS, Xout);
    half_t* tmp = Xin; Xin = Xout; Xout = tmp;
  }

  // --- prediction head (co=48 -> 3 co tiles, stride 48) --------------------
  conv_wmma<64,1,3,48><<<dim3(ROWS*4),32,0,stream>>>(Xin, pPred, P(80), bufG, 1);
  att_logits<<<(ROWS*TT)/256, 256, 0, stream>>>(Xin, P(3), P(8), attL);
  row_reduce<<<ROWS, 256, 0, stream>>>(bufG, attL, pa);
  head_out<<<(ROWS*32 + 255)/256, 256, 0, stream>>>(
      pa, P(4), P(79), P(6), P(81), (float*)d_out);
}
